// MoeWNA16Matmul_71141838291320
// MI455X (gfx1250) — matrix operand order repaired and checked by reading, not compiled, not run on hardware
//
#include <hip/hip_runtime.h>

// Problem constants (from reference)
constexpr int E_    = 8;
constexpr int T_    = 8192;
constexpr int KDIM  = 2048;
constexpr int NDIM  = 2048;
constexpr int G_    = 16;

// Tiling
constexpr int BM   = 64;            // tokens per sorted block
constexpr int BN   = 128;           // N columns per workgroup (8 waves x 16)
constexpr int KC   = 128;           // K chunk staged in LDS
constexpr int MAXB = T_ / BM + E_;  // 136 max padded blocks
constexpr int MAXP = MAXB * BM;     // 8704 sorted-id slots

typedef __attribute__((ext_vector_type(16))) __bf16        v16bf;
typedef __attribute__((ext_vector_type(2)))  __bf16        v2bf;
typedef __attribute__((ext_vector_type(8)))  float         v8f;
typedef __attribute__((ext_vector_type(8)))  unsigned int  v8u;

// ---------------------------------------------------------------------------
// Pass 1: counting-sort tokens by expert, pad each expert to a multiple of BM.
// Single workgroup. sorted[] pads are -1.
// ---------------------------------------------------------------------------
__global__ void moe_align_kernel(const int* __restrict__ eids,
                                 int* __restrict__ sorted,
                                 int* __restrict__ blk_e,
                                 int* __restrict__ nblk) {
  __shared__ int cnt[E_];
  __shared__ int base[E_];
  __shared__ int fill[E_];
  const int tid = threadIdx.x;
  if (tid < E_) { cnt[tid] = 0; fill[tid] = 0; }
  __syncthreads();
  for (int t = tid; t < T_; t += blockDim.x) atomicAdd(&cnt[eids[t]], 1);
  __syncthreads();
  if (tid == 0) {
    int off = 0;
    for (int e = 0; e < E_; ++e) {
      base[e] = off;
      int padded = ((cnt[e] + BM - 1) / BM) * BM;
      for (int b = off / BM; b < (off + padded) / BM; ++b) blk_e[b] = e;
      off += padded;
    }
    *nblk = off / BM;
  }
  __syncthreads();
  for (int i = tid; i < MAXP; i += blockDim.x) sorted[i] = -1;
  __syncthreads();
  for (int t = tid; t < T_; t += blockDim.x) {
    int e = eids[t];
    int pos = base[e] + atomicAdd(&fill[e], 1);
    sorted[pos] = t;
  }
}

// ---------------------------------------------------------------------------
// Pass 2: per block of 64 sorted tokens (single expert) x 128 N columns.
// 8 waves; each wave: 16 N cols, 4 M-tiles, bf16 WMMA with f32 accum.
// int4 weights dequantized in-register; one B fragment feeds 4 WMMAs.
// ---------------------------------------------------------------------------
__global__ void __launch_bounds__(256)
moe_wna16_gemm_kernel(const float* __restrict__ state,
                      const unsigned int* __restrict__ wp,
                      const float* __restrict__ scale,
                      const int* __restrict__ zp,
                      const int* __restrict__ g_idx,
                      const int* __restrict__ sorted,
                      const int* __restrict__ blk_e,
                      const int* __restrict__ nblk,
                      float* __restrict__ out) {
  const int b = blockIdx.x;
  if (b >= *nblk) return;

  __shared__ __align__(16) __bf16 a_lds[BM * KC];  // 16 KB bf16 A tile
  __shared__ float2 lut[G_ * BN];                  // 16 KB (s, s*zp)
  __shared__ int    g_lds[KC];
  __shared__ int    ids[BM];

  const int tid   = threadIdx.x;
  const int e     = blk_e[b];
  const int nbase = blockIdx.y * BN;

  // Stage token ids and the per-(group, n) dequant LUT once per workgroup.
  if (tid < BM) ids[tid] = sorted[b * BM + tid];
  for (int idx = tid; idx < G_ * BN; idx += 256) {
    int g  = idx >> 7;          // idx / BN
    int nl = idx & (BN - 1);    // idx % BN
    size_t off = ((size_t)e * G_ + g) * NDIM + nbase + nl;
    float s = scale[off];
    float z = (float)zp[off];
    lut[g * BN + nl] = make_float2(s, s * z);
  }

  const int lane = tid & 31;
  const int wave = tid >> 5;
  const int hf   = lane >> 4;   // lane half (0/1)
  const int lm   = lane & 15;
  const int nl   = wave * 16 + lm;       // local N column (0..127)
  const int ng   = nbase + nl;           // global N column
  const unsigned int* wpe = wp + (size_t)e * (KDIM / 8) * NDIM + ng;

  v8f acc[4] = {};  // 4 M-tiles of 16x16 f32

  for (int kc0 = 0; kc0 < KDIM; kc0 += KC) {
    __syncthreads();  // ids/lut ready; previous-iter reads of a_lds done
    if (tid < KC) g_lds[tid] = g_idx[kc0 + tid];
    // Stage A (f32 -> bf16), 4 elements per thread iteration:
    // global_load_b128 + packed bf16 convert + ds_store_b64.
    for (int idx = tid; idx < BM * KC / 4; idx += 256) {
      int i4 = idx * 4;
      int m  = i4 >> 7;          // i4 / KC
      int kk = i4 & (KC - 1);    // i4 % KC
      int t  = ids[m];
      float4 v = (t >= 0) ? *(const float4*)&state[(size_t)t * KDIM + kc0 + kk]
                          : make_float4(0.f, 0.f, 0.f, 0.f);
      v2bf lo = {(__bf16)v.x, (__bf16)v.y};
      v2bf hi = {(__bf16)v.z, (__bf16)v.w};
      uint2 packed = make_uint2(__builtin_bit_cast(unsigned int, lo),
                                __builtin_bit_cast(unsigned int, hi));
      *(uint2*)&a_lds[i4] = packed;
    }
    if (kc0 + KC < KDIM)  // gfx1250 global_prefetch of next weight chunk
      __builtin_prefetch(wpe + (size_t)((kc0 + KC) >> 3) * NDIM, 0, 1);
    __syncthreads();

#pragma unroll
    for (int kb = 0; kb < KC; kb += 32) {
      // ---- B fragment: 16 consecutive K at (kc0+kb+16*hf) for column ng ----
      const int kpg = (kc0 + kb + 16 * hf) >> 3;          // packed-dword row
      const unsigned int wlo = wpe[(size_t)kpg * NDIM];
      const unsigned int whi = wpe[(size_t)(kpg + 1) * NDIM];
      v16bf bfrag;
#pragma unroll
      for (int j = 0; j < 8; ++j) {
        unsigned int w = (j < 4) ? wlo : whi;
        int sh = 8 * (j & 3);
        float q0 = (float)((w >> sh) & 15u);
        float q1 = (float)((w >> (sh + 4)) & 15u);
        int kl = kb + 16 * hf + 2 * j;
        float2 sz0 = lut[g_lds[kl] * BN + nl];
        float2 sz1 = lut[g_lds[kl + 1] * BN + nl];
        bfrag[2 * j]     = (__bf16)(q0 * sz0.x - sz0.y);   // (q - zp) * s
        bfrag[2 * j + 1] = (__bf16)(q1 * sz1.x - sz1.y);
      }

      // ---- 4 WMMAs reuse the same B fragment (amortize dequant VALU) ----
#pragma unroll
      for (int mt = 0; mt < 4; ++mt) {
        const __bf16* arow = &a_lds[(mt * 16 + lm) * KC + kb];
        uint4 x = *(const uint4*)(arow + 16 * hf);       // K = kb+16h .. +7
        uint4 y = *(const uint4*)(arow + 16 * hf + 8);   // K = kb+16h+8 .. +7
        v8u ab = {x.x, x.y, x.z, x.w, y.x, y.y, y.z, y.w};
        v16bf afrag = __builtin_bit_cast(v16bf, ab);
        acc[mt] = __builtin_amdgcn_wmma_f32_16x16x32_bf16(
            false, afrag, false, bfrag, (short)0, acc[mt], false, false);
      }
    }
  }

  // Epilogue: C/D layout row = r + 8*hf within each 16x16 tile.
#pragma unroll
  for (int mt = 0; mt < 4; ++mt) {
#pragma unroll
    for (int r = 0; r < 8; ++r) {
      int m = mt * 16 + r + 8 * hf;
      int t = ids[m];
      if (t >= 0) out[(size_t)t * NDIM + ng] = acc[mt][r];
    }
  }
}

// ---------------------------------------------------------------------------
extern "C" void kernel_launch(void* const* d_in, const int* in_sizes, int n_in,
                              void* d_out, int out_size, void* d_ws, size_t ws_size,
                              hipStream_t stream) {
  const float*        state = (const float*)d_in[0];
  const unsigned int* wpack = (const unsigned int*)d_in[1];
  const float*        scl   = (const float*)d_in[2];
  const int*          zpt   = (const int*)d_in[3];
  const int*          gidx  = (const int*)d_in[4];
  const int*          eids  = (const int*)d_in[5];
  float*              out   = (float*)d_out;

  int* sorted = (int*)d_ws;
  int* blk_e  = sorted + MAXP;
  int* nblk   = blk_e + MAXB;

  moe_align_kernel<<<1, 256, 0, stream>>>(eids, sorted, blk_e, nblk);

  dim3 grid(MAXB, NDIM / BN);
  moe_wna16_gemm_kernel<<<grid, 256, 0, stream>>>(
      state, wpack, scl, zpt, gidx, sorted, blk_e, nblk, out);
}
